// ScaledDotProductAttention_73443940761807
// MI455X (gfx1250) — matrix operand order repaired and checked by reading, not compiled, not run on hardware
//
#include <hip/hip_runtime.h>
#include <hip/hip_bf16.h>

// ---------------------------------------------------------------------------
// Problem constants (B=8, S=2048, F=1024 from the reference setup_inputs)
// ---------------------------------------------------------------------------
#define B_DIM 8
#define S_DIM 2048
#define F_DIM 1024
#define BK    32      // K-step per WMMA sweep (16x16x32 bf16)
#define AST   40      // LDS row stride (ushort elems) for 32-wide staging tiles
#define KST   1032    // LDS row stride (ushort elems) for resident K block

typedef __bf16          v16bf __attribute__((ext_vector_type(16)));
typedef float           v8f   __attribute__((ext_vector_type(8)));
typedef unsigned short  us_t;
typedef us_t            u16x8 __attribute__((ext_vector_type(8)));

// fp32 -> bf16 with round-to-nearest-even
__device__ __forceinline__ us_t f2bf(float f) {
  unsigned u = __float_as_uint(f);
  unsigned r = (u + 0x7FFFu + ((u >> 16) & 1u)) >> 16;
  return (us_t)r;
}

__device__ __forceinline__ u16x8 pack8(float4 a, float4 b) {
  u16x8 r;
  r[0] = f2bf(a.x); r[1] = f2bf(a.y); r[2] = f2bf(a.z); r[3] = f2bf(a.w);
  r[4] = f2bf(b.x); r[5] = f2bf(b.y); r[6] = f2bf(b.z); r[7] = f2bf(b.w);
  return r;
}

union FragU { v16bf v; u16x8 p[2]; };

// A fragment (16x32 bf16, M x K): lane (h*16+m) holds K = {8h..8h+7, 16+8h..16+8h+7}
__device__ __forceinline__ v16bf frag_a(const us_t* rowp, int h) {
  FragU u;
  u.p[0] = *(const u16x8*)(rowp + 8 * h);
  u.p[1] = *(const u16x8*)(rowp + 16 + 8 * h);
  return u.v;
}
// B fragment (32x16 bf16, K x N) for B = (row-major tile)^T:
// lane (h*16+n) holds K = {8h..8h+7, 16+8h..16+8h+7} from the tile row n
__device__ __forceinline__ v16bf frag_b(const us_t* rowp, int h) {
  FragU u;
  u.p[0] = *(const u16x8*)(rowp + 8 * h);
  u.p[1] = *(const u16x8*)(rowp + 16 + 8 * h);
  return u.v;
}

// load 16 consecutive fp32 from global and convert to 16 bf16
__device__ __forceinline__ void fetch16(const float* __restrict__ base,
                                        int row, int col, u16x8& lo, u16x8& hi) {
  const float4* p = (const float4*)(base + (size_t)row * F_DIM + col);
  float4 a = p[0], b = p[1], c = p[2], d = p[3];
  lo = pack8(a, b);
  hi = pack8(c, d);
}

// ---- CDNA5 async memory->LDS (ASYNCcnt-tracked, no VGPR data path) --------
// GV mode: vdst = LDS byte offset (wave-relative), vaddr = 64-bit global addr.
__device__ __forceinline__ void async_ld_b128(unsigned lds_byte_off, const void* gaddr) {
  asm volatile("global_load_async_to_lds_b128 %0, %1, off"
               :: "v"(lds_byte_off), "v"(gaddr)
               : "memory");
}
__device__ __forceinline__ void async_wait0() {
  asm volatile("s_wait_asynccnt 0x0" ::: "memory");
}

// ---------------------------------------------------------------------------
// Kernel A: Y[r,g] = sum_f X[r,f] * W[g,f]   (X: [16384,1024] fp32, W: [1024,1024] fp32)
// Output bf16 (Q_t, K_t) or fp32 (V_t). 128x128 tile, 8 waves, BK=32 pipeline.
// fp32->bf16 conversion happens in the register staging path, so no async here.
// ---------------------------------------------------------------------------
template <bool OUT_BF16>
__global__ __launch_bounds__(256)
void proj_gemm(const float* __restrict__ X, const float* __restrict__ W,
               us_t* __restrict__ Ybf, float* __restrict__ Yf) {
  __shared__ us_t As[2][128 * AST];
  __shared__ us_t Bs[2][128 * AST];

  const int tid     = threadIdx.x;
  const int rowBase = blockIdx.y * 128;
  const int colBase = blockIdx.x * 128;

  const int srow = tid >> 1;          // staging row 0..127
  const int scol = (tid & 1) * 16;    // staging col 0 or 16 within BK=32

  const int lane = tid & 31;
  const int w    = tid >> 5;
  const int wm   = w & 3;             // wave M index (0..3), 32 rows each
  const int wn   = w >> 2;            // wave N index (0..1), 64 cols each
  const int h    = lane >> 4;
  const int ml   = lane & 15;

  v8f acc[2][4] = {};

  u16x8 aLo, aHi, bLo, bHi;
  // prologue: stage k0 = 0 into buffer 0
  fetch16(X, rowBase + srow, scol, aLo, aHi);
  fetch16(W, colBase + srow, scol, bLo, bHi);
  *(u16x8*)(&As[0][srow * AST + scol])     = aLo;
  *(u16x8*)(&As[0][srow * AST + scol + 8]) = aHi;
  *(u16x8*)(&Bs[0][srow * AST + scol])     = bLo;
  *(u16x8*)(&Bs[0][srow * AST + scol + 8]) = bHi;
  __syncthreads();

  int buf = 0;
  const int NK = F_DIM / BK;  // 32
  for (int kk = 0; kk < NK; ++kk) {
    const bool hasNext = (kk + 1 < NK);
    if (hasNext) {  // issue global loads early; they overlap the WMMAs below
      fetch16(X, rowBase + srow, (kk + 1) * BK + scol, aLo, aHi);
      fetch16(W, colBase + srow, (kk + 1) * BK + scol, bLo, bHi);
    }
    v16bf af[2], bfv[4];
#pragma unroll
    for (int fm = 0; fm < 2; ++fm)
      af[fm] = frag_a(&As[buf][(wm * 32 + fm * 16 + ml) * AST], h);
#pragma unroll
    for (int fn = 0; fn < 4; ++fn)
      bfv[fn] = frag_b(&Bs[buf][(wn * 64 + fn * 16 + ml) * AST], h);
#pragma unroll
    for (int fm = 0; fm < 2; ++fm)
#pragma unroll
      for (int fn = 0; fn < 4; ++fn)
        acc[fm][fn] = __builtin_amdgcn_wmma_f32_16x16x32_bf16(
            false, af[fm], false, bfv[fn], (short)0, acc[fm][fn], false, false);
    if (hasNext) {
      const int nb = buf ^ 1;
      *(u16x8*)(&As[nb][srow * AST + scol])     = aLo;
      *(u16x8*)(&As[nb][srow * AST + scol + 8]) = aHi;
      *(u16x8*)(&Bs[nb][srow * AST + scol])     = bLo;
      *(u16x8*)(&Bs[nb][srow * AST + scol + 8]) = bHi;
    }
    __syncthreads();
    buf ^= 1;
  }

  // epilogue: D element r lives at (m = r + 8h, n = ml) of each 16x16 frag
#pragma unroll
  for (int fm = 0; fm < 2; ++fm)
#pragma unroll
    for (int fn = 0; fn < 4; ++fn)
#pragma unroll
      for (int r = 0; r < 8; ++r) {
        const int gm = rowBase + wm * 32 + fm * 16 + r + 8 * h;
        const int gn = colBase + wn * 64 + fn * 16 + ml;
        const float vv = acc[fm][fn][r];
        if (OUT_BF16) Ybf[(size_t)gm * F_DIM + gn] = f2bf(vv);
        else          Yf [(size_t)gm * F_DIM + gn] = vv;
      }
}

// ---------------------------------------------------------------------------
// Kernel B: per (batch b, 128-column block kb) own the softmax(axis=1) column
// statistics. K block (128 x 1024 bf16 = 258KB) is DMA'd into LDS with
// GLOBAL_LOAD_ASYNC_TO_LDS_B128 and stays resident (WGP LDS = 320KB); Q tiles
// stream through an async double buffer. Deterministic fixed-order reductions.
// Emits factor[b,s] = exp(a[s,s]) / sum_i exp(a[i,s]).
// ---------------------------------------------------------------------------
__global__ __launch_bounds__(256)
void attn_diag_softmax(const us_t* __restrict__ qt, const us_t* __restrict__ kt,
                       float* __restrict__ factor) {
  extern __shared__ char smem[];
  us_t*  Ks     = (us_t*)smem;                                     // 128*KST
  us_t*  Qs     = (us_t*)(smem + (size_t)128 * KST * 2);           // 2*128*AST
  float* part   = (float*)(smem + (size_t)128 * KST * 2 + (size_t)2 * 128 * AST * 2);
  float* colsum = part + 128 * 8;
  float* dexp   = colsum + 128;

  // LDS byte offsets (wave-relative) for the async engine; dynamic LDS begins
  // right after the (empty) static segment.
  const unsigned ldsBase = __builtin_amdgcn_groupstaticsize();
  const unsigned KsOff   = ldsBase;
  const unsigned QsOff   = ldsBase + 128u * KST * 2u;

  const int tid   = threadIdx.x;
  const int kb    = blockIdx.x;      // 0..15
  const int b     = blockIdx.y;      // 0..7
  const int kBase = kb * 128;
  const size_t batchOff = (size_t)b * S_DIM * F_DIM;

  // stage resident K block: 128 rows x 1024 bf16 via async memory->LDS
  {
    const int row  = tid >> 1;
    const int half = tid & 1;
    const us_t* gsrc   = kt + batchOff + (size_t)(kBase + row) * F_DIM + half * 512;
    const unsigned dst = KsOff + (unsigned)(row * KST + half * 512) * 2u;
#pragma unroll 8
    for (int c = 0; c < 64; ++c)
      async_ld_b128(dst + (unsigned)c * 16u, gsrc + c * 8);
  }
  if (tid < 128) { colsum[tid] = 0.f; dexp[tid] = 0.f; }
  async_wait0();
  __syncthreads();

  const int lane = tid & 31;
  const int w    = tid >> 5;
  const int wm   = w & 3;
  const int wn   = w >> 2;
  const int h    = lane >> 4;
  const int ml   = lane & 15;
  const int srow = tid >> 1;
  const int scol = (tid & 1) * 16;
  const float scl = 0.03125f;  // 1/sqrt(1024)

  for (int ib = 0; ib < S_DIM / 128; ++ib) {
    const int iBase = ib * 128;
    v8f acc[2][4] = {};

    const us_t*    qrow   = qt + batchOff + (size_t)(iBase + srow) * F_DIM + scol;
    const unsigned qdst0  = QsOff + (unsigned)((srow * AST + scol) * 2);

    // prologue: async-stage Q tile (k0 = 0) into buffer 0
    async_ld_b128(qdst0,       qrow);
    async_ld_b128(qdst0 + 16u, qrow + 8);
    async_wait0();
    __syncthreads();

    int buf = 0;
    for (int kk = 0; kk < F_DIM / BK; ++kk) {
      const bool hasNext = (kk + 1 < F_DIM / BK);
      if (hasNext) {  // fill the other buffer while WMMAs run below
        const unsigned d = qdst0 + (unsigned)((buf ^ 1) * 128 * AST * 2);
        const us_t*    s = qrow + (kk + 1) * BK;
        async_ld_b128(d,       s);
        async_ld_b128(d + 16u, s + 8);
      }
      v16bf af[2], bfv[4];
#pragma unroll
      for (int fm = 0; fm < 2; ++fm)
        af[fm] = frag_a(Qs + buf * 128 * AST + (wm * 32 + fm * 16 + ml) * AST, h);
#pragma unroll
      for (int fn = 0; fn < 4; ++fn)
        bfv[fn] = frag_b(Ks + (wn * 64 + fn * 16 + ml) * KST + kk * BK, h);
#pragma unroll
      for (int fm = 0; fm < 2; ++fm)
#pragma unroll
        for (int fn = 0; fn < 4; ++fn)
          acc[fm][fn] = __builtin_amdgcn_wmma_f32_16x16x32_bf16(
              false, af[fm], false, bfv[fn], (short)0, acc[fm][fn], false, false);
      if (hasNext) async_wait0();
      __syncthreads();
      buf ^= 1;
    }

    // epilogue: e = exp(score*scale); per-column partials + diagonal capture
    float psum[4] = {0.f, 0.f, 0.f, 0.f};
#pragma unroll
    for (int fm = 0; fm < 2; ++fm)
#pragma unroll
      for (int fn = 0; fn < 4; ++fn)
#pragma unroll
        for (int r = 0; r < 8; ++r) {
          const float e = __expf(acc[fm][fn][r] * scl);
          psum[fn] += e;
          const int gi = iBase + wm * 32 + fm * 16 + r + 8 * h;   // global row i
          const int gn = wn * 64 + fn * 16 + ml;                  // local col
          if (gi == kBase + gn) dexp[gn] = e;                     // unique writer
        }
#pragma unroll
    for (int fn = 0; fn < 4; ++fn)
      part[(wn * 64 + fn * 16 + ml) * 8 + wm * 2 + h] = psum[fn];
    __syncthreads();
    if (tid < 128) {  // fixed-order (deterministic) reduction
      float s = 0.f;
#pragma unroll
      for (int j = 0; j < 8; ++j) s += part[tid * 8 + j];
      colsum[tid] += s;
    }
    __syncthreads();
  }

  if (tid < 128)
    factor[(size_t)b * S_DIM + kBase + tid] = dexp[tid] / colsum[tid];
}

// ---------------------------------------------------------------------------
// Kernel C: out[b,s,:] = factor[b,s] * V_t[b,s,:]   (float4 vectorized)
// ---------------------------------------------------------------------------
__global__ __launch_bounds__(256)
void scale_v(const float* __restrict__ vt, const float* __restrict__ factor,
             float* __restrict__ out) {
  const size_t i4  = (size_t)blockIdx.x * blockDim.x + threadIdx.x;  // float4 idx
  const size_t row = i4 >> 8;  // 256 float4 per 1024-wide row
  const float f = factor[row];
  float4 v = ((const float4*)vt)[i4];
  float4 o = make_float4(v.x * f, v.y * f, v.z * f, v.w * f);
  ((float4*)out)[i4] = o;
}

// ---------------------------------------------------------------------------
extern "C" void kernel_launch(void* const* d_in, const int* in_sizes, int n_in,
                              void* d_out, int out_size, void* d_ws, size_t ws_size,
                              hipStream_t stream) {
  const float* q  = (const float*)d_in[0];
  const float* k  = (const float*)d_in[1];
  const float* v  = (const float*)d_in[2];
  const float* Wq = (const float*)d_in[3];
  const float* Wk = (const float*)d_in[4];
  const float* Wv = (const float*)d_in[5];

  const size_t nBSF = (size_t)B_DIM * S_DIM * F_DIM;  // 16,777,216
  char* ws = (char*)d_ws;
  us_t*  qt = (us_t*)ws;    ws += nBSF * 2;   // Q_t  bf16  32MB
  us_t*  kt = (us_t*)ws;    ws += nBSF * 2;   // K_t  bf16  32MB
  float* vt = (float*)ws;   ws += nBSF * 4;   // V_t  fp32  64MB
  float* fac = (float*)ws;                    // factor     64KB

  dim3 gGemm(F_DIM / 128, (B_DIM * S_DIM) / 128);  // (8, 128)
  proj_gemm<true ><<<gGemm, 256, 0, stream>>>(q, Wq, qt, nullptr);
  proj_gemm<true ><<<gGemm, 256, 0, stream>>>(k, Wk, kt, nullptr);
  proj_gemm<false><<<gGemm, 256, 0, stream>>>(v, Wv, nullptr, vt);

  const size_t smemB = (size_t)128 * KST * 2   // resident K block (258KB)
                     + (size_t)2 * 128 * AST * 2
                     + 128 * 8 * 4 + 128 * 4 + 128 * 4;
  hipFuncSetAttribute(reinterpret_cast<const void*>(&attn_diag_softmax),
                      hipFuncAttributeMaxDynamicSharedMemorySize, (int)smemB);
  attn_diag_softmax<<<dim3(S_DIM / 128, B_DIM), 256, smemB, stream>>>(qt, kt, fac);

  scale_v<<<(unsigned)((nBSF / 4) / 256), 256, 0, stream>>>(vt, fac, (float*)d_out);
}
